// HeteroGNN_40432822124774
// MI455X (gfx1250) — compile-verified, matrix operand order checked by reading, not run on hardware
//
#include <hip/hip_runtime.h>
#include <hip/hip_bf16.h>

// ---------------- problem constants (match reference) ----------------
#define N_W 500000
#define N_V 50000
#define N_H 5000
#define E1  500000
#define E2  50000
#define HC  128
#define OUTC 8
#define NG  512

typedef __attribute__((ext_vector_type(16))) __bf16 v16bf;
typedef __attribute__((ext_vector_type(8)))  float  v8f;

// ---------------------------------------------------------------------
// Weight convert: W[k][n] f32 row-major -> Wt[n][k] bf16 (N-major so a
// B-fragment's K-run is contiguous -> 16B vector loads).
__global__ void cvt_w_kernel(const float* __restrict__ W, __bf16* __restrict__ Wt) {
    int i = blockIdx.x * blockDim.x + threadIdx.x;   // 0..16383
    if (i >= HC * HC) return;
    int k = i >> 7, n = i & 127;
    Wt[n * HC + k] = (__bf16)W[k * HC + n];
}

// ---------------------------------------------------------------------
// Degree kernels (float atomics; counts are small integers).
__global__ void deg2_kernel(const int* __restrict__ src, const int* __restrict__ dst,
                            float* __restrict__ deg_s, float* __restrict__ deg_d, int E) {
    int e = blockIdx.x * blockDim.x + threadIdx.x;
    if (e < E) {
        atomicAdd(&deg_s[src[e]], 1.0f);
        atomicAdd(&deg_d[dst[e]], 1.0f);
    }
}
__global__ void deg1_kernel(const int* __restrict__ dst, float* __restrict__ deg, int E) {
    int e = blockIdx.x * blockDim.x + threadIdx.x;
    if (e < E) atomicAdd(&deg[dst[e]], 1.0f);
}

// ---------------------------------------------------------------------
// GCN edge scatter (linearity rewrite: aggregate x BEFORE the matmul).
// agg1 += coef * x[src]       (feeds vm1 GEMM)
// agg2 += coef * relu(x)[src] (feeds vm2 GEMM)
__global__ void gcn_scatter_kernel(const float* __restrict__ x,
                                   const int* __restrict__ src, const int* __restrict__ dst,
                                   const float* __restrict__ deg_s, const float* __restrict__ deg_d,
                                   float* __restrict__ agg1, float* __restrict__ agg2, int E) {
    int c = threadIdx.x;                       // 0..127 column
    for (int e = blockIdx.x; e < E; e += gridDim.x) {
        int s = src[e], d = dst[e];
        float ds = deg_s[s], dd = deg_d[d];
        float inv_s = ds > 0.0f ? rsqrtf(ds) : 0.0f;   // deg>0 => max(deg,1)==deg
        float inv_d = dd > 0.0f ? rsqrtf(dd) : 0.0f;
        float coef = inv_s * inv_d;
        float v = x[(size_t)s * HC + c];
        atomicAdd(&agg1[(size_t)d * HC + c], v * coef);
        atomicAdd(&agg2[(size_t)d * HC + c], fmaxf(v, 0.0f) * coef);
    }
}

// SAGE neighbor-sum scatter: agg[dst] += xsrc[src]
__global__ void sage_scatter_kernel(const float* __restrict__ xsrc,
                                    const int* __restrict__ src, const int* __restrict__ dst,
                                    float* __restrict__ agg, int E) {
    int c = threadIdx.x;
    for (int e = blockIdx.x; e < E; e += gridDim.x) {
        int s = src[e], d = dst[e];
        atomicAdd(&agg[(size_t)d * HC + c], xsrc[(size_t)s * HC + c]);
    }
}

// ---------------------------------------------------------------------
// Pool: segment-sum of relu(x) over sorted batch ids; run-length flush
// so atomics fire roughly once per graph boundary per block.
#define POOL_ROWS 512
__global__ void pool_kernel(const float* __restrict__ x, const int* __restrict__ batch,
                            float* __restrict__ poolacc, float* __restrict__ cnt, int NW) {
    int c = threadIdx.x;                 // 0..127
    int r0 = blockIdx.x * POOL_ROWS;
    if (r0 >= NW) return;
    int rend = min(r0 + POOL_ROWS, NW);
    float accum = 0.0f, count = 0.0f;
    int gcur = batch[r0];
    for (int r = r0; r < rend; ++r) {
        int g = batch[r];
        if (g != gcur) {
            atomicAdd(&poolacc[(size_t)gcur * HC + c], accum);
            if (c == 0) atomicAdd(&cnt[gcur], count);
            accum = 0.0f; count = 0.0f; gcur = g;
        }
        accum += fmaxf(x[(size_t)r * HC + c], 0.0f);
        count += 1.0f;
    }
    atomicAdd(&poolacc[(size_t)gcur * HC + c], accum);
    if (c == 0) atomicAdd(&cnt[gcur], count);
}

// ---------------------------------------------------------------------
// WMMA GEMM:  C[M,128] = act( s1(row)*A1 @ B1  +  [A2 @ B2]  + bias )
// bf16 operands, f32 accumulate via v_wmma_f32_16x16x32_bf16.
// One wave per 16-row tile; 8 col-tiles x 4 K-steps.
// A-frag (16-bit 16x32): lane<16 holds row=lane, K {k0..k0+7, k0+16..k0+23};
// lane>=16 holds row=lane-16, K {k0+8..k0+15, k0+24..k0+31}.
// B-frag mirrors with N in place of M (Bt stored N-major).
__device__ inline v16bf load_a_frag(const float* __restrict__ A, int row, int kA, float scale) {
    const float* p = A + (size_t)row * HC + kA;
    v16bf f;
#pragma unroll
    for (int i = 0; i < 8; ++i) f[i]     = (__bf16)(p[i] * scale);
#pragma unroll
    for (int i = 0; i < 8; ++i) f[8 + i] = (__bf16)(p[16 + i] * scale);
    return f;
}
__device__ inline v16bf load_b_frag(const __bf16* __restrict__ Bt, int n, int kB) {
    const __bf16* p = Bt + (size_t)n * HC + kB;   // contiguous K => 2x16B loads
    v16bf f;
#pragma unroll
    for (int i = 0; i < 8; ++i) f[i]     = p[i];
#pragma unroll
    for (int i = 0; i < 8; ++i) f[8 + i] = p[16 + i];
    return f;
}

__global__ void gemm_wmma_kernel(const float* __restrict__ A1, const float* __restrict__ scale1,
                                 const __bf16* __restrict__ B1t,
                                 const float* __restrict__ A2, const __bf16* __restrict__ B2t,
                                 const float* __restrict__ bias,
                                 float* __restrict__ C, int M, int relu_flag) {
    int tile = blockIdx.x * (blockDim.x >> 5) + (threadIdx.x >> 5);  // wave-uniform
    int ntiles = (M + 15) >> 4;
    if (tile >= ntiles) return;                    // uniform: EXEC stays all-1 for live waves
    int lane = threadIdx.x & 31;
    int halfsel = lane >> 4;
    int koff = halfsel ? 8 : 0;
    int rloc = lane & 15;
    int row = tile * 16 + rloc;
    int rowc = min(row, M - 1);
    bool full_tile = (tile * 16 + 16) <= M;        // wave-uniform: unguarded stores

    float s1 = 1.0f;
    if (scale1) s1 = 1.0f / fmaxf(scale1[rowc], 1.0f);   // SAGE mean

    v16bf a1[4], a2[4];
#pragma unroll
    for (int ks = 0; ks < 4; ++ks)
        a1[ks] = load_a_frag(A1, rowc, ks * 32 + koff, s1);
    if (A2) {
#pragma unroll
        for (int ks = 0; ks < 4; ++ks)
            a2[ks] = load_a_frag(A2, rowc, ks * 32 + koff, 1.0f);
    }

    for (int ct = 0; ct < 8; ++ct) {
        int n = ct * 16 + rloc;
        v8f acc = {0.f, 0.f, 0.f, 0.f, 0.f, 0.f, 0.f, 0.f};

        // Batch all 4 B fragments first (one load clause), then chain WMMAs
        // back-to-back so fetch overlaps the matrix pipe.
        v16bf b1f[4];
#pragma unroll
        for (int ks = 0; ks < 4; ++ks) b1f[ks] = load_b_frag(B1t, n, ks * 32 + koff);
#pragma unroll
        for (int ks = 0; ks < 4; ++ks)
            acc = __builtin_amdgcn_wmma_f32_16x16x32_bf16(false, a1[ks], false, b1f[ks],
                                                          (short)0, acc, false, false);
        if (A2) {
            v16bf b2f[4];
#pragma unroll
            for (int ks = 0; ks < 4; ++ks) b2f[ks] = load_b_frag(B2t, n, ks * 32 + koff);
#pragma unroll
            for (int ks = 0; ks < 4; ++ks)
                acc = __builtin_amdgcn_wmma_f32_16x16x32_bf16(false, a2[ks], false, b2f[ks],
                                                              (short)0, acc, false, false);
        }

        // C/D layout: VGPR r -> (M = r + 8*halfsel, N = ct*16 + rloc)
        float bn = bias[n];
        if (full_tile) {
#pragma unroll
            for (int r = 0; r < 8; ++r) {
                int grow = tile * 16 + r + koff;            // koff==8*halfsel
                float v = acc[r] + bn;
                if (relu_flag) v = fmaxf(v, 0.0f);
                C[(size_t)grow * HC + n] = v;
            }
        } else {
#pragma unroll
            for (int r = 0; r < 8; ++r) {
                int grow = tile * 16 + r + koff;
                if (grow < M) {
                    float v = acc[r] + bn;
                    if (relu_flag) v = fmaxf(v, 0.0f);
                    C[(size_t)grow * HC + n] = v;
                }
            }
        }
    }
}

// ---------------------------------------------------------------------
// Final FC: out[g][o] = (poolacc[g]/max(cnt,1)) . W_fc[:,o] + b_fc[o]
__global__ void fc_kernel(const float* __restrict__ poolacc, const float* __restrict__ cnt,
                          const float* __restrict__ Wfc, const float* __restrict__ bfc,
                          float* __restrict__ out) {
    int i = blockIdx.x * blockDim.x + threadIdx.x;     // G*OUT = 4096
    if (i >= NG * OUTC) return;
    int g = i >> 3, o = i & 7;
    float inv = 1.0f / fmaxf(cnt[g], 1.0f);
    float s = bfc[o];
    for (int c = 0; c < HC; ++c) s += poolacc[(size_t)g * HC + c] * inv * Wfc[c * OUTC + o];
    out[i] = s;
}

// ---------------------------------------------------------------------
extern "C" void kernel_launch(void* const* d_in, const int* in_sizes, int n_in,
                              void* d_out, int out_size, void* d_ws, size_t ws_size,
                              hipStream_t stream) {
    const float* x_w     = (const float*)d_in[0];
    const float* x_vm    = (const float*)d_in[1];
    const float* x_host  = (const float*)d_in[2];
    const int*   s_w2v   = (const int*)d_in[3];
    const int*   d_w2v   = (const int*)d_in[4];
    const int*   s_v2h   = (const int*)d_in[5];
    const int*   d_v2h   = (const int*)d_in[6];
    const int*   wbatch  = (const int*)d_in[7];
    const float* W_gcn1  = (const float*)d_in[8];
    const float* b_gcn1  = (const float*)d_in[9];
    const float* Wl1     = (const float*)d_in[10];
    const float* bl1     = (const float*)d_in[11];
    const float* Wr1     = (const float*)d_in[12];
    const float* W_gcn2  = (const float*)d_in[13];
    const float* b_gcn2  = (const float*)d_in[14];
    const float* Wl2     = (const float*)d_in[15];
    const float* bl2     = (const float*)d_in[16];
    const float* Wr2     = (const float*)d_in[17];
    const float* W_fc    = (const float*)d_in[18];
    const float* b_fc    = (const float*)d_in[19];

    float* out_fc   = (float*)d_out;                                   // [512,8]
    float* out_vm2  = (float*)d_out + (size_t)NG * OUTC;               // [50000,128]
    float* out_host2= out_vm2 + (size_t)N_V * HC;                      // [5000,128]

    // ---- workspace layout ----
    char* ws = (char*)d_ws;
    size_t off = 0;
    auto take = [&](size_t bytes) { size_t o = off; off = (off + bytes + 255) & ~(size_t)255; return o; };
    size_t o_wt      = take(6 * (size_t)HC * HC * sizeof(__bf16));     // 6 bf16 weight mats
    size_t zero_beg  = off;
    size_t o_deg_s   = take((size_t)N_W * 4);
    size_t o_deg_d   = take((size_t)N_V * 4);
    size_t o_deg_h   = take((size_t)N_H * 4);
    size_t o_vmagg1  = take((size_t)N_V * HC * 4);
    size_t o_vmagg2  = take((size_t)N_V * HC * 4);
    size_t o_hagg1   = take((size_t)N_H * HC * 4);
    size_t o_hagg2   = take((size_t)N_H * HC * 4);
    size_t o_pool    = take((size_t)NG * HC * 4);
    size_t o_cnt     = take((size_t)NG * 4);
    size_t zero_end  = off;
    size_t o_vm1     = take((size_t)N_V * HC * 4);   // fully overwritten, no zeroing
    size_t o_host1   = take((size_t)N_H * HC * 4);

    __bf16* wt_gcn1 = (__bf16*)(ws + o_wt) + 0 * HC * HC;
    __bf16* wt_gcn2 = (__bf16*)(ws + o_wt) + 1 * HC * HC;
    __bf16* wt_l1   = (__bf16*)(ws + o_wt) + 2 * HC * HC;
    __bf16* wt_r1   = (__bf16*)(ws + o_wt) + 3 * HC * HC;
    __bf16* wt_l2   = (__bf16*)(ws + o_wt) + 4 * HC * HC;
    __bf16* wt_r2   = (__bf16*)(ws + o_wt) + 5 * HC * HC;
    float* deg_s  = (float*)(ws + o_deg_s);
    float* deg_d  = (float*)(ws + o_deg_d);
    float* deg_h  = (float*)(ws + o_deg_h);
    float* vmagg1 = (float*)(ws + o_vmagg1);
    float* vmagg2 = (float*)(ws + o_vmagg2);
    float* hagg1  = (float*)(ws + o_hagg1);
    float* hagg2  = (float*)(ws + o_hagg2);
    float* poolac = (float*)(ws + o_pool);
    float* cnt    = (float*)(ws + o_cnt);
    float* vm1    = (float*)(ws + o_vm1);
    float* host1  = (float*)(ws + o_host1);

    // ---- zero accumulators (graph-capturable memset) ----
    hipMemsetAsync(ws + zero_beg, 0, zero_end - zero_beg, stream);

    // ---- weight convert/transpose to bf16 ----
    dim3 cb(256), cg((HC * HC + 255) / 256);
    cvt_w_kernel<<<cg, cb, 0, stream>>>(W_gcn1, wt_gcn1);
    cvt_w_kernel<<<cg, cb, 0, stream>>>(W_gcn2, wt_gcn2);
    cvt_w_kernel<<<cg, cb, 0, stream>>>(Wl1,    wt_l1);
    cvt_w_kernel<<<cg, cb, 0, stream>>>(Wr1,    wt_r1);
    cvt_w_kernel<<<cg, cb, 0, stream>>>(Wl2,    wt_l2);
    cvt_w_kernel<<<cg, cb, 0, stream>>>(Wr2,    wt_r2);

    // ---- degrees ----
    deg2_kernel<<<(E1 + 255) / 256, 256, 0, stream>>>(s_w2v, d_w2v, deg_s, deg_d, E1);
    deg1_kernel<<<(E2 + 255) / 256, 256, 0, stream>>>(d_v2h, deg_h, E2);

    // ---- edge aggregation (x pre-matmul; linearity rewrite) ----
    gcn_scatter_kernel<<<E1, HC, 0, stream>>>(x_w, s_w2v, d_w2v, deg_s, deg_d, vmagg1, vmagg2, E1);
    sage_scatter_kernel<<<E2, HC, 0, stream>>>(x_vm, s_v2h, d_v2h, hagg1, E2);
    pool_kernel<<<(N_W + POOL_ROWS - 1) / POOL_ROWS, HC, 0, stream>>>(x_w, wbatch, poolac, cnt, N_W);

    // ---- WMMA GEMMs (256 threads = 8 waves/block, 1 tile/wave) ----
    int vm_tiles = (N_V + 15) / 16, vm_blocks = (vm_tiles + 7) / 8;
    int h_tiles  = (N_H + 15) / 16, h_blocks  = (h_tiles + 7) / 8;
    // vm1 = relu(vmagg1 @ Wgcn1 + b1)
    gemm_wmma_kernel<<<vm_blocks, 256, 0, stream>>>(vmagg1, nullptr, wt_gcn1,
                                                    nullptr, nullptr, b_gcn1, vm1, N_V, 1);
    // vm2 = vmagg2 @ Wgcn2 + b2   -> output
    gemm_wmma_kernel<<<vm_blocks, 256, 0, stream>>>(vmagg2, nullptr, wt_gcn2,
                                                    nullptr, nullptr, b_gcn2, out_vm2, N_V, 0);
    // host1 = relu((hagg1/deg) @ Wl1 + x_host @ Wr1 + bl1)
    gemm_wmma_kernel<<<h_blocks, 256, 0, stream>>>(hagg1, deg_h, wt_l1,
                                                   x_host, wt_r1, bl1, host1, N_H, 1);
    // second SAGE aggregation uses vm1
    sage_scatter_kernel<<<E2, HC, 0, stream>>>(vm1, s_v2h, d_v2h, hagg2, E2);
    // host2 = (hagg2/deg) @ Wl2 + host1 @ Wr2 + bl2 -> output
    gemm_wmma_kernel<<<h_blocks, 256, 0, stream>>>(hagg2, deg_h, wt_l2,
                                                   host1, wt_r2, bl2, out_host2, N_H, 0);

    // ---- final FC ----
    fc_kernel<<<(NG * OUTC + 255) / 256, 256, 0, stream>>>(poolac, cnt, W_fc, b_fc, out_fc);
}
